// DRMM_84971632984330
// MI455X (gfx1250) — matrix operand order repaired and checked by reading, not compiled, not run on hardware
//
#include <hip/hip_runtime.h>
#include <hip/hip_bf16.h>

typedef __attribute__((ext_vector_type(16))) __bf16 v16bf;
typedef __attribute__((ext_vector_type(8)))  __bf16 v8bf;
typedef __attribute__((ext_vector_type(8)))  float  v8f;

#define EMB   300
#define EPAD  320
#define NBINS 5
#define BN    32
#define QN    16
#define DN    10
#define LN    1000
#define TD    64                          // doc terms per LDS tile
#define NT    ((LN + TD - 1) / TD)        // 16 tiles
#define KSTEPS (EPAD / 32)                // 10 WMMA K-steps

static __device__ __forceinline__ unsigned short f2bf(float f) {
    unsigned int u = __float_as_uint(f);
    u += 0x7FFFu + ((u >> 16) & 1u);      // round-to-nearest-even
    return (unsigned short)(u >> 16);
}

// ---------------------------------------------------------------------------
// Kernel 1: per batch, normalize query embeddings -> bf16 (padded to 320),
// compute gate logits and softmax term weights.  16 waves, one per query.
// ---------------------------------------------------------------------------
__global__ void __launch_bounds__(512)
drmm_prep_q(const int* __restrict__ queries, const float* __restrict__ emb,
            const float* __restrict__ gate_w, const float* __restrict__ gate_b,
            unsigned short* __restrict__ qEmb, float* __restrict__ tw) {
    int b    = blockIdx.x;
    int q    = threadIdx.x >> 5;
    int lane = threadIdx.x & 31;
    __shared__ float sLogit[QN];

    int qidx = queries[b * QN + q];
    const float* row = emb + (long)qidx * EMB;

    float x[10];
    float s2 = 0.f, gl = 0.f;
#pragma unroll
    for (int i = 0; i < 10; ++i) {
        int k   = lane + 32 * i;
        int kc  = (k < EMB) ? k : 0;
        float v = row[kc];
        float g = gate_w[kc];
        v = (k < EMB) ? v : 0.f;
        x[i] = v;
        s2  += v * v;
        gl  += v * g;
    }
#pragma unroll
    for (int off = 16; off; off >>= 1) {
        s2 += __shfl_xor(s2, off, 32);
        gl += __shfl_xor(gl, off, 32);
    }
    float rn = (s2 > 0.f) ? rsqrtf(s2) : 0.f;

    unsigned short* dst = qEmb + ((long)b * QN + q) * EPAD;
#pragma unroll
    for (int i = 0; i < 10; ++i)
        dst[lane + 32 * i] = f2bf(x[i] * rn);

    if (lane == 0) sLogit[q] = gl + gate_b[0];
    __syncthreads();

    if (threadIdx.x == 0) {
        float mx = sLogit[0];
        for (int i = 1; i < QN; ++i) mx = fmaxf(mx, sLogit[i]);
        float e[QN], s = 0.f;
        for (int i = 0; i < QN; ++i) { e[i] = __expf(sLogit[i] - mx); s += e[i]; }
        float inv = 1.f / s;
        for (int i = 0; i < QN; ++i) tw[b * QN + i] = e[i] * inv;
    }
}

// ---------------------------------------------------------------------------
// Kernel 2: one block per (b,d).  Gather/normalize 64 doc terms to LDS bf16
// (prefetching the next tile's rows), WMMA 16 queries x 16 terms per wave
// with A fragments in registers and double-buffered B fragments, branchless
// w1-binning of each cosine, reduce, tiny FFNN + gated pooling.
// ---------------------------------------------------------------------------
__global__ void __launch_bounds__(128)
drmm_score(const int* __restrict__ docs, const float* __restrict__ emb,
           const unsigned short* __restrict__ qEmb, const float* __restrict__ tw,
           const float* __restrict__ pw1, const float* __restrict__ pb1,
           const float* __restrict__ pw2, const float* __restrict__ pb2,
           const float* __restrict__ p_ow, const float* __restrict__ p_ob,
           float* __restrict__ out) {
    int bd   = blockIdx.x;
    int b    = bd / DN;
    int d    = bd % DN;
    int wave = threadIdx.x >> 5;          // 0..3
    int lane = threadIdx.x & 31;
    int m    = lane & 15;
    int half = lane >> 4;

    __shared__ unsigned short sQ[QN * EPAD];   // 10 KB
    __shared__ unsigned short sD[TD * EPAD];   // 40 KB
    __shared__ float sAcc[QN];
    __shared__ float sW1[NBINS];

    // stage normalized bf16 query tile (16x320) into LDS, 16B vector copies
    {
        const uint4* src4 = (const uint4*)(qEmb + (long)b * QN * EPAD);
        uint4*       dst4 = (uint4*)sQ;
        for (int i = threadIdx.x; i < (QN * EPAD * 2) / 16; i += 128)
            dst4[i] = src4[i];
    }
    if (threadIdx.x < QN)    sAcc[threadIdx.x] = 0.f;
    if (threadIdx.x < NBINS) sW1[threadIdx.x]  = pw1[threadIdx.x];
    __syncthreads();

    // ---- A fragments (queries) are tile-invariant: hoist into registers ----
    // lane L: row m = L&15, K chunks [(L>>4)*8 .. +7] and [+16 .. +23]
    const unsigned short* qrow = sQ + m * EPAD + half * 8;
    v16bf afrag[KSTEPS];
#pragma unroll
    for (int kk = 0; kk < KSTEPS; ++kk) {
        v8bf alo = *(const v8bf*)(qrow + kk * 32);
        v8bf ahi = *(const v8bf*)(qrow + kk * 32 + 16);
        afrag[kk] = __builtin_shufflevector(alo, ahi,
                        0,1,2,3,4,5,6,7,8,9,10,11,12,13,14,15);
    }

    float acc[8];
#pragma unroll
    for (int r = 0; r < 8; ++r) acc[r] = 0.f;

    const unsigned short* drow = sD + (wave * 16 + m) * EPAD + half * 8;
    const long docBase = ((long)b * DN + d) * LN;

    auto loadB = [&](int kk) -> v16bf {
        v8bf blo = *(const v8bf*)(drow + kk * 32);
        v8bf bhi = *(const v8bf*)(drow + kk * 32 + 16);
        return __builtin_shufflevector(blo, bhi,
                   0,1,2,3,4,5,6,7,8,9,10,11,12,13,14,15);
    };

    for (int tile = 0; tile < NT; ++tile) {
        int base = tile * TD;

        // ---- gather + normalize 64 doc-term embeddings (wave per term) ----
        for (int t = 0; t < TD / 4; ++t) {
            int j = wave + 4 * t;
            int l = base + j;

            // prefetch next tile's row for this (wave, t) slot into caches
            int lp = l + TD;
            if (lp < LN) {
                int idxp = docs[docBase + lp];
                const char* prow = (const char*)(emb + (long)idxp * EMB);
                __builtin_prefetch(prow + lane * 40, 0, 0);  // global_prefetch_b8
            }

            float x[10];
            float s2 = 0.f;
            if (l < LN) {
                int idx = docs[docBase + l];
                const float* row = emb + (long)idx * EMB;
#pragma unroll
                for (int i = 0; i < 10; ++i) {
                    int k   = lane + 32 * i;
                    int kc  = (k < EMB) ? k : 0;
                    float v = row[kc];
                    v = (k < EMB) ? v : 0.f;
                    x[i] = v;
                    s2  += v * v;
                }
#pragma unroll
                for (int off = 16; off; off >>= 1)
                    s2 += __shfl_xor(s2, off, 32);
            } else {
#pragma unroll
                for (int i = 0; i < 10; ++i) x[i] = 0.f;
            }
            float rn = (s2 > 0.f) ? rsqrtf(s2) : 0.f;
            unsigned short* dst = sD + j * EPAD;
#pragma unroll
            for (int i = 0; i < 10; ++i)
                dst[lane + 32 * i] = f2bf(x[i] * rn);
        }
        __syncthreads();

        // ---- WMMA: this wave's 16 doc terms vs all 16 queries ----
        // double-buffered B fragments so ds_load latency overlaps the WMMA
        v8f c = {};
        v16bf bcur = loadB(0);
#pragma unroll
        for (int kk = 0; kk < KSTEPS; ++kk) {
            v16bf bnext = (kk + 1 < KSTEPS) ? loadB(kk + 1) : bcur;
            c = __builtin_amdgcn_wmma_f32_16x16x32_bf16(
                    false, afrag[kk], false, bcur, (short)0, c, false, false);
            bcur = bnext;
        }

        // ---- branchless binning through w1 (per query M = r + 8*half) ----
        bool lanevalid = (base + wave * 16 + m) < LN;   // this lane's term index
#pragma unroll
        for (int r = 0; r < 8; ++r) {
            float cv = c[r];
            int ix = (int)((cv + 1.f) * 2.f);
            ix = ix < 0 ? 0 : (ix > 4 ? 4 : ix);
            float wv = sW1[ix];                          // unconditional ds_load
            bool in = lanevalid && (cv >= -1.f) && (cv <= 1.f);
            acc[r] += in ? wv : 0.f;                     // v_cndmask, no branch
        }
        __syncthreads();   // protect sD before next tile's gather
    }

    // ---- reduce over the 16 lanes (N) of each half, combine waves in LDS ----
#pragma unroll
    for (int r = 0; r < 8; ++r) {
        float v = acc[r];
#pragma unroll
        for (int off = 8; off; off >>= 1)
            v += __shfl_xor(v, off, 16);
        if (m == 0) atomicAdd(&sAcc[half * 8 + r], v);
    }
    __syncthreads();

    if (threadIdx.x == 0) {
        float vb1 = pb1[0], vw2 = pw2[0], vb2 = pb2[0];
        float pooled = 0.f;
        for (int q2 = 0; q2 < QN; ++q2) {
            float h = sAcc[q2] + vb1;          // hist . w1 + b1
            float f = h * vw2 + vb2;           // second linear
            pooled += f * tw[b * QN + q2];     // gated pooling
        }
        out[bd] = pooled * p_ow[0] + p_ob[0];
    }
}

// ---------------------------------------------------------------------------
extern "C" void kernel_launch(void* const* d_in, const int* in_sizes, int n_in,
                              void* d_out, int out_size, void* d_ws, size_t ws_size,
                              hipStream_t stream) {
    const int*   queries = (const int*)  d_in[0];
    const int*   docs    = (const int*)  d_in[1];
    const float* emb     = (const float*)d_in[2];
    const float* gate_w  = (const float*)d_in[3];
    const float* gate_b  = (const float*)d_in[4];
    const float* w1      = (const float*)d_in[5];
    const float* b1      = (const float*)d_in[6];
    const float* w2      = (const float*)d_in[7];
    const float* b2      = (const float*)d_in[8];
    const float* ow      = (const float*)d_in[9];
    const float* ob      = (const float*)d_in[10];
    float*       out     = (float*)d_out;

    unsigned short* qEmb = (unsigned short*)d_ws;
    float* tw = (float*)((char*)d_ws + (size_t)BN * QN * EPAD * sizeof(unsigned short));

    drmm_prep_q<<<BN, 512, 0, stream>>>(queries, emb, gate_w, gate_b, qEmb, tw);
    drmm_score<<<BN * DN, 128, 0, stream>>>(docs, emb, qEmb, tw,
                                            w1, b1, w2, b2, ow, ob, out);
}
